// MultiHead_Self_Attention_56770877719113
// MI455X (gfx1250) — matrix operand order repaired and checked by reading, not compiled, not run on hardware
//
#include <hip/hip_runtime.h>

typedef __bf16 bf16_t;
typedef __attribute__((ext_vector_type(16))) __bf16 v16bf;
typedef __attribute__((ext_vector_type(8)))  __bf16 v8bf;
typedef __attribute__((ext_vector_type(8)))  float  v8f;

#define NB 4
#define NS 2048
#define ND 1024
#define NH 16
#define DK 64

// ---------------------------------------------------------------------------
// WMMA helpers (gfx1250 wave32 layouts per CDNA5 ISA 7.12.2)
// ---------------------------------------------------------------------------

__device__ inline v8f wmma_bf16(v16bf a, v16bf b, v8f c) {
  // D = A(16x32 bf16) * B(32x16 bf16) + C(16x16 f32)
  return __builtin_amdgcn_wmma_f32_16x16x32_bf16(
      /*neg_a=*/false, a, /*neg_b=*/false, b,
      /*c_mod=*/(short)0, c, /*reuse_a=*/false, /*reuse_b=*/false);
}

// A fragment 16x32 (MxK): row m = lane%16; element j -> K = j + 16*hi
// lane<16 : K 0-15;  lane>=16: K 16-31 (matches load_b)
// Two groups of 8 contiguous bf16 (16B) -> two b128 loads.
__device__ inline v16bf load_a(const bf16_t* base, int ld, int koff, int lane) {
  int m = lane & 15, hi = lane >> 4;
  const bf16_t* row = base + (size_t)m * ld + koff + 16 * hi;
  v8bf g0 = *(const v8bf*)(row);
  v8bf g1 = *(const v8bf*)(row + 8);
  return __builtin_shufflevector(g0, g1, 0,1,2,3,4,5,6,7,8,9,10,11,12,13,14,15);
}

// B fragment 32x16 (KxN): col n = lane%16; element j -> K = koff + j + 16*hi.
// Source stored row-major as src[n][k] (i.e. B is src transposed), so the 16
// K-values per lane are contiguous -> one 32B vector load.
__device__ inline v16bf load_b(const bf16_t* base, int ld, int koff, int lane) {
  int n = lane & 15, hi = lane >> 4;
  return *(const v16bf*)(base + (size_t)n * ld + koff + 16 * hi);
}

// Row reductions across the 16 lanes holding one C-fragment row.
__device__ inline float rowmax16(float v) {
  v = fmaxf(v, __shfl_xor(v, 1, 32));
  v = fmaxf(v, __shfl_xor(v, 2, 32));
  v = fmaxf(v, __shfl_xor(v, 4, 32));
  v = fmaxf(v, __shfl_xor(v, 8, 32));
  return v;
}
__device__ inline float rowsum16(float v) {
  v += __shfl_xor(v, 1, 32);
  v += __shfl_xor(v, 2, 32);
  v += __shfl_xor(v, 4, 32);
  v += __shfl_xor(v, 8, 32);
  return v;
}

// ---------------------------------------------------------------------------
// Kernel 1: fp32 -> bf16 cast
// ---------------------------------------------------------------------------
__global__ void cast_bf16_kernel(const float* __restrict__ src,
                                 bf16_t* __restrict__ dst, int n) {
  int i = blockIdx.x * blockDim.x + threadIdx.x;
  int stride = gridDim.x * blockDim.x;
  for (; i < n; i += stride) dst[i] = (bf16_t)src[i];
}

// ---------------------------------------------------------------------------
// Kernel 2: QKV projection. One wave computes a 64x32 super-tile of X * W^T
// (4 M-subtiles x 2 N-subtiles register-blocked: 12 b128 loads / 8 WMMAs).
// mode 0/1: RoPE + write [B,H,S,DK] (Q / K). mode 2: write V^T [B,H,DK,S].
// ---------------------------------------------------------------------------
__global__ __launch_bounds__(256) void proj_kernel(
    const bf16_t* __restrict__ X, const bf16_t* __restrict__ W,
    bf16_t* __restrict__ dst, const int* __restrict__ pos, int mode) {
  int lane = threadIdx.x & 31;
  int wave = blockIdx.x * 8 + (threadIdx.x >> 5);   // 4096 waves
  int mt = wave >> 5;                                // 128 super row tiles (64 rows)
  int nt = wave & 31;                                // 32 super col tiles (32 cols)
  int m0 = mt * 64, n0 = nt * 32;

  v8f acc[4][2] = {};
  const bf16_t* Xrow = X + (size_t)m0 * ND;
  const bf16_t* Wrow = W + (size_t)n0 * ND;
#pragma unroll 2
  for (int kk = 0; kk < ND; kk += 32) {
    v16bf a[4], b[2];
#pragma unroll
    for (int mi = 0; mi < 4; mi++)
      a[mi] = load_a(Xrow + (size_t)mi * 16 * ND, ND, kk, lane);
#pragma unroll
    for (int ni = 0; ni < 2; ni++)
      b[ni] = load_b(Wrow + (size_t)ni * 16 * ND, ND, kk, lane);
#pragma unroll
    for (int mi = 0; mi < 4; mi++)
#pragma unroll
      for (int ni = 0; ni < 2; ni++)
        acc[mi][ni] = wmma_bf16(a[mi], b[ni], acc[mi][ni]);
  }

  int n = lane & 15, hi = lane >> 4;
#pragma unroll
  for (int ni = 0; ni < 2; ni++) {
    int o = n0 + ni * 16 + n;     // output channel
    int h = o >> 6;               // head
    int dh = o & 63;              // dim within head
    if (mode < 2) {
      // RoPE: pair (2t, 2t+1) lives in adjacent lanes (adjacent n).
      float invf = __powf(10000.f, -(float)(dh & ~1) * (1.0f / 64.0f));
#pragma unroll
      for (int mi = 0; mi < 4; mi++) {
#pragma unroll
        for (int r = 0; r < 8; r++) {
          int sg = m0 + mi * 16 + r + 8 * hi;   // flat row in [B*S]
          int bb = sg >> 11;                    // / NS
          int s = sg & (NS - 1);
          float ang = (float)pos[s] * invf;
          float c = __cosf(ang), sn = __sinf(ang);
          float val = acc[mi][ni][r];
          float part = __shfl_xor(val, 1, 32);
          float outv = (o & 1) ? (part * sn + val * c)   // x1' = x0*sin + x1*cos
                               : (val * c - part * sn);  // x0' = x0*cos - x1*sin
          dst[(((size_t)bb * NH + h) * NS + s) * DK + dh] = (bf16_t)outv;
        }
      }
    } else {
      // V transposed: [B,H,DK,S] so attention's PV B-fragments are contiguous.
#pragma unroll
      for (int mi = 0; mi < 4; mi++) {
#pragma unroll
        for (int r = 0; r < 8; r++) {
          int sg = m0 + mi * 16 + r + 8 * hi;
          int bb = sg >> 11;
          int s = sg & (NS - 1);
          dst[(((size_t)bb * NH + h) * DK + dh) * NS + s] = (bf16_t)acc[mi][ni][r];
        }
      }
    }
  }
}

// ---------------------------------------------------------------------------
// Kernel 3: causal flash attention. One wave per (b, h, 32-row Q tile).
// 32 keys/step: K- and V-fragments are shared across both 16-row Q subtiles.
// ---------------------------------------------------------------------------
__global__ __launch_bounds__(256) void attn_kernel(
    const bf16_t* __restrict__ Qb, const bf16_t* __restrict__ Kb,
    const bf16_t* __restrict__ Vt, bf16_t* __restrict__ AO) {
  __shared__ __align__(16) bf16_t psh[8][32 * 32];   // per-wave P staging (2 KB)
  int lane = threadIdx.x & 31;
  int wslot = threadIdx.x >> 5;
  int wave = blockIdx.x * 8 + wslot;                 // 4096 waves
  int qt = wave & 63;                                // S/32 tiles
  int h = (wave >> 6) & 15;
  int b = wave >> 10;
  int q0 = qt * 32;

  const bf16_t* Qrow = Qb + (((size_t)b * NH + h) * NS + q0) * DK;
  const bf16_t* Krow = Kb + (((size_t)b * NH + h) * NS) * DK;
  const bf16_t* Vrow = Vt + (((size_t)b * NH + h) * DK) * NS;

  // Two 16-row Q subtiles resident in registers for the whole K loop.
  v16bf qa[2][2];
#pragma unroll
  for (int u = 0; u < 2; u++) {
    qa[u][0] = load_a(Qrow + (size_t)u * 16 * DK, DK, 0, lane);
    qa[u][1] = load_a(Qrow + (size_t)u * 16 * DK, DK, 32, lane);
  }

  int n = lane & 15, hi = lane >> 4;
  float mrow[2][8], lrow[2][8];
  v8f oacc[2][4] = {};
#pragma unroll
  for (int u = 0; u < 2; u++)
#pragma unroll
    for (int r = 0; r < 8; r++) { mrow[u][r] = -1e30f; lrow[u][r] = 0.f; }
  const float sc = 0.125f;   // 1/sqrt(64)

  bf16_t* P = &psh[wslot][0];
  int kbmax = (q0 + 31) >> 5;
  for (int kb = 0; kb <= kbmax; kb++) {
    int k0 = kb * 32;
    // ---- K fragments, shared by both Q subtiles ----
    v16bf kb0 = load_b(Krow + (size_t)k0 * DK, DK, 0, lane);
    v16bf kb1 = load_b(Krow + (size_t)k0 * DK, DK, 32, lane);
    v16bf kb2 = load_b(Krow + (size_t)(k0 + 16) * DK, DK, 0, lane);
    v16bf kb3 = load_b(Krow + (size_t)(k0 + 16) * DK, DK, 32, lane);
#pragma unroll
    for (int u = 0; u < 2; u++) {
      v8f s0 = {}, s1 = {};
      s0 = wmma_bf16(qa[u][0], kb0, s0);
      s0 = wmma_bf16(qa[u][1], kb1, s0);
      s1 = wmma_bf16(qa[u][0], kb2, s1);
      s1 = wmma_bf16(qa[u][1], kb3, s1);
      // ---- online softmax + stage P (bf16, 16x32) to LDS ----
#pragma unroll
      for (int r = 0; r < 8; r++) {
        int q = q0 + u * 16 + r + 8 * hi;
        float v0 = s0[r] * sc, v1 = s1[r] * sc;
        if (k0 + n > q) v0 = -1e30f;            // causal mask
        if (k0 + 16 + n > q) v1 = -1e30f;
        float tm = rowmax16(fmaxf(v0, v1));
        float nm = fmaxf(mrow[u][r], tm);
        float fs = __expf(mrow[u][r] - nm);
        float p0 = __expf(v0 - nm), p1 = __expf(v1 - nm);
        lrow[u][r] = lrow[u][r] * fs + rowsum16(p0 + p1);
        mrow[u][r] = nm;
        oacc[u][0][r] *= fs; oacc[u][1][r] *= fs;
        oacc[u][2][r] *= fs; oacc[u][3][r] *= fs;
        P[(u * 16 + r + 8 * hi) * 32 + n] = (bf16_t)p0;
        P[(u * 16 + r + 8 * hi) * 32 + 16 + n] = (bf16_t)p1;
      }
    }
    asm volatile("s_wait_dscnt 0x0" ::: "memory");
    // ---- P(32x32) * V(32x64): V fragments shared by both Q subtiles ----
    v16bf pa0 = load_a(P, 32, 0, lane);
    v16bf pa1 = load_a(P + 16 * 32, 32, 0, lane);
#pragma unroll
    for (int vt = 0; vt < 4; vt++) {
      v16bf bv = load_b(Vrow + (size_t)vt * 16 * NS, NS, k0, lane);
      oacc[0][vt] = wmma_bf16(pa0, bv, oacc[0][vt]);
      oacc[1][vt] = wmma_bf16(pa1, bv, oacc[1][vt]);
    }
  }

  // ---- normalize and write attn-out in [B,S,H*DK] layout (bf16) ----
#pragma unroll
  for (int u = 0; u < 2; u++) {
#pragma unroll
    for (int r = 0; r < 8; r++) {
      int q = q0 + u * 16 + r + 8 * hi;
      float invl = 1.0f / lrow[u][r];
      size_t rowbase = ((size_t)b * NS + q) * ND + (size_t)h * DK;
      AO[rowbase + 0 + n]  = (bf16_t)(oacc[u][0][r] * invl);
      AO[rowbase + 16 + n] = (bf16_t)(oacc[u][1][r] * invl);
      AO[rowbase + 32 + n] = (bf16_t)(oacc[u][2][r] * invl);
      AO[rowbase + 48 + n] = (bf16_t)(oacc[u][3][r] * invl);
    }
  }
}

// ---------------------------------------------------------------------------
// Kernel 4: output projection (64x32 register-blocked), fp32 result.
// ---------------------------------------------------------------------------
__global__ __launch_bounds__(256) void outproj_kernel(
    const bf16_t* __restrict__ AOin, const bf16_t* __restrict__ Wo,
    float* __restrict__ out) {
  int lane = threadIdx.x & 31;
  int wave = blockIdx.x * 8 + (threadIdx.x >> 5);   // 4096 waves
  int mt = wave >> 5, nt = wave & 31;
  int m0 = mt * 64, n0 = nt * 32;

  v8f acc[4][2] = {};
  const bf16_t* Arow = AOin + (size_t)m0 * ND;
  const bf16_t* Wrow = Wo + (size_t)n0 * ND;
#pragma unroll 2
  for (int kk = 0; kk < ND; kk += 32) {
    v16bf a[4], b[2];
#pragma unroll
    for (int mi = 0; mi < 4; mi++)
      a[mi] = load_a(Arow + (size_t)mi * 16 * ND, ND, kk, lane);
#pragma unroll
    for (int ni = 0; ni < 2; ni++)
      b[ni] = load_b(Wrow + (size_t)ni * 16 * ND, ND, kk, lane);
#pragma unroll
    for (int mi = 0; mi < 4; mi++)
#pragma unroll
      for (int ni = 0; ni < 2; ni++)
        acc[mi][ni] = wmma_bf16(a[mi], b[ni], acc[mi][ni]);
  }
  int n = lane & 15, hi = lane >> 4;
#pragma unroll
  for (int mi = 0; mi < 4; mi++)
#pragma unroll
    for (int ni = 0; ni < 2; ni++)
#pragma unroll
      for (int r = 0; r < 8; r++) {
        int m = mi * 16 + r + 8 * hi;
        out[(size_t)(m0 + m) * ND + n0 + ni * 16 + n] = acc[mi][ni][r];
      }
}

// ---------------------------------------------------------------------------
extern "C" void kernel_launch(void* const* d_in, const int* in_sizes, int n_in,
                              void* d_out, int out_size, void* d_ws, size_t ws_size,
                              hipStream_t stream) {
  const float* x  = (const float*)d_in[0];
  const int* pos  = (const int*)d_in[1];
  const float* wq = (const float*)d_in[2];
  const float* wk = (const float*)d_in[3];
  const float* wv = (const float*)d_in[4];
  const float* wo = (const float*)d_in[5];
  float* out = (float*)d_out;

  // Workspace layout (bf16 elements). Total 44M elems = 88 MB.
  bf16_t* w = (bf16_t*)d_ws;
  const size_t M1 = 1024u * 1024u;
  bf16_t* Xbf = w;                 // 8M  : x cast
  bf16_t* Wqb = w + 8 * M1;        // 1M each
  bf16_t* Wkb = w + 9 * M1;
  bf16_t* Wvb = w + 10 * M1;
  bf16_t* Wob = w + 11 * M1;
  bf16_t* Qb  = w + 12 * M1;       // 8M : [B,H,S,DK] + RoPE
  bf16_t* Kb  = w + 20 * M1;       // 8M : [B,H,S,DK] + RoPE
  bf16_t* Vt  = w + 28 * M1;       // 8M : [B,H,DK,S] (transposed)
  bf16_t* AO  = w + 36 * M1;       // 8M : attn output [B,S,D]

  cast_bf16_kernel<<<2048, 256, 0, stream>>>(x, Xbf, NB * NS * ND);
  cast_bf16_kernel<<<256, 256, 0, stream>>>(wq, Wqb, ND * ND);
  cast_bf16_kernel<<<256, 256, 0, stream>>>(wk, Wkb, ND * ND);
  cast_bf16_kernel<<<256, 256, 0, stream>>>(wv, Wvb, ND * ND);
  cast_bf16_kernel<<<256, 256, 0, stream>>>(wo, Wob, ND * ND);

  // 128 x 32 super-tiles, 1 wave each, 8 waves/block -> 512 blocks
  proj_kernel<<<512, 256, 0, stream>>>(Xbf, Wqb, Qb, pos, 0);
  proj_kernel<<<512, 256, 0, stream>>>(Xbf, Wkb, Kb, pos, 1);
  proj_kernel<<<512, 256, 0, stream>>>(Xbf, Wvb, Vt, pos, 2);

  // B*H*(S/32) = 4096 waves -> 512 blocks
  attn_kernel<<<512, 256, 0, stream>>>(Qb, Kb, Vt, AO);

  outproj_kernel<<<512, 256, 0, stream>>>(AO, Wob, out);
}
